// GraphAttention_25512105738328
// MI455X (gfx1250) — matrix operand order repaired and checked by reading, not compiled, not run on hardware
//
#include <hip/hip_runtime.h>
#include <math.h>
#include <stdint.h>

typedef __bf16 bf16_t;
typedef __attribute__((ext_vector_type(16))) __bf16 v16bf;
typedef __attribute__((ext_vector_type(8)))  float  v8f;
typedef __attribute__((ext_vector_type(4)))  unsigned int u32x4;
typedef __attribute__((ext_vector_type(8)))  int i32x8;
typedef __attribute__((ext_vector_type(4)))  int i32x4;

#define NEG_MAX (-3.402823466e38f)

// ---------------- WMMA fragment helpers (contiguous k order, same for A and B) ----------------
// A (16x32 bf16): lane m=lane%16 is row M; lane-half hh selects K bytes:
//   K = kofs + hh*16 + e
static __device__ __forceinline__ v16bf load_a_bf16(const bf16_t* base, int rstride,
                                                    int kofs, int mrows) {
  const int lane = threadIdx.x & 31;
  const int m = lane & 15, hh = lane >> 4;
  v16bf a;
  if (m < mrows) {
    const bf16_t* p = base + m * rstride + kofs + hh * 16;
#pragma unroll
    for (int e = 0; e < 8; ++e) a[e] = p[e];
#pragma unroll
    for (int e = 0; e < 8; ++e) a[8 + e] = p[8 + e];
  } else {
#pragma unroll
    for (int e = 0; e < 16; ++e) a[e] = (bf16_t)0.0f;
  }
  return a;
}

// A fragment from f32 source (LDS-resident TDM tile), converting to bf16 inline.
static __device__ __forceinline__ v16bf load_a_f32(const float* base, int rstride,
                                                   int kofs, int mrows) {
  const int lane = threadIdx.x & 31;
  const int m = lane & 15, hh = lane >> 4;
  v16bf a;
  if (m < mrows) {
    const float* p = base + m * rstride + kofs + hh * 16;
#pragma unroll
    for (int e = 0; e < 8; ++e) a[e] = (bf16_t)p[e];
#pragma unroll
    for (int e = 0; e < 8; ++e) a[8 + e] = (bf16_t)p[8 + e];
  } else {
#pragma unroll
    for (int e = 0; e < 16; ++e) a[e] = (bf16_t)0.0f;
  }
  return a;
}

// B (32x16 bf16) from row-major KxN bf16 source.
static __device__ __forceinline__ v16bf load_b_bf16(const bf16_t* base, int rstride) {
  const int lane = threadIdx.x & 31;
  const int nn = lane & 15, hh = lane >> 4;
  const bf16_t* p = base + (hh * 16) * rstride + nn;
  v16bf b;
#pragma unroll
  for (int e = 0; e < 16; ++e) b[e] = p[e * rstride];
  return b;
}

// B (32x16 bf16) = transpose of a row-major NxK bf16 source (B[k][n] = src[n][k]).
static __device__ __forceinline__ v16bf load_bT_bf16(const bf16_t* base, int rstride) {
  const int lane = threadIdx.x & 31;
  const int nn = lane & 15, hh = lane >> 4;
  const bf16_t* p = base + nn * rstride + hh * 16;
  v16bf b;
#pragma unroll
  for (int e = 0; e < 16; ++e) b[e] = p[e];
  return b;
}

// B (32x16 bf16) from row-major f32 KxN source with column guard (zero-pad N).
static __device__ __forceinline__ v16bf load_b_f32(const float* base, int rstride, int ncols) {
  const int lane = threadIdx.x & 31;
  const int nn = lane & 15, hh = lane >> 4;
  v16bf b;
#pragma unroll
  for (int e = 0; e < 16; ++e)
    b[e] = (nn < ncols) ? (bf16_t)base[(hh * 16 + e) * rstride + nn] : (bf16_t)0.0f;
  return b;
}

static __device__ __forceinline__ v8f wmma_bf16(v16bf a, v16bf b, v8f c) {
  return __builtin_amdgcn_wmma_f32_16x16x32_bf16(false, a, false, b, (short)0, c,
                                                 false, false);
}

// ---------------- Tensor Data Mover: 2-D f32 tile -> LDS with row padding ----------------
// D# packing per CDNA5 ISA. Moves tile_rows x 64 f32 elements,
// inserting 1 DWORD of LDS padding after every 64 DWORDs (row stride 65 in LDS).
static __device__ __forceinline__ void tdm_load_rows_f32(
    unsigned lds_byte_off, const float* gptr,
    unsigned tile_rows, unsigned tensor_rows, unsigned row_stride_elems) {
  unsigned long long ga = (unsigned long long)(uintptr_t)gptr;
  u32x4 g0;
  g0[0] = 1u;                                               // count=1, user-mode, no gather
  g0[1] = lds_byte_off;                                     // lds_addr
  g0[2] = (unsigned)(ga & 0xffffffffu);                     // global_addr[31:0]
  g0[3] = (unsigned)((ga >> 32) & 0x01ffffffu) | (2u << 30);// global_addr[56:32] | type=2

  const unsigned tensor_d = 64, tile_d = 64;
  i32x8 g1;
  g1[0] = (int)((2u << 16)      // data_size = 4B
              | (1u << 20)      // pad_enable
              | (5u << 22)      // pad_interval: 64 DWORDs
              | (0u << 25));    // pad_amount: 1 DWORD
  g1[1] = (int)((tensor_d & 0xffffu) << 16);                         // tensor_dim0[15:0]
  g1[2] = (int)((tensor_d >> 16) | ((tensor_rows & 0xffffu) << 16)); // dim0 hi | dim1 lo
  g1[3] = (int)((tensor_rows >> 16) | (tile_d << 16));               // dim1 hi | tile_dim0
  g1[4] = (int)(tile_rows & 0xffffu);                                // tile_dim1 | tile_dim2=0
  g1[5] = (int)row_stride_elems;                                     // tensor_dim0_stride
  g1[6] = 0;
  g1[7] = 0;

  i32x4 z4 = {0, 0, 0, 0};
#if defined(__clang_major__) && (__clang_major__ >= 23)
  i32x8 z8 = {0, 0, 0, 0, 0, 0, 0, 0};
  __builtin_amdgcn_tensor_load_to_lds(g0, g1, z4, z4, z8, 0);
#else
  __builtin_amdgcn_tensor_load_to_lds(g0, g1, z4, z4, 0);
#endif
}

// =============== Kernel 1: QKV projection (WMMA bf16) + rotary ===============
__global__ __launch_bounds__(256) void k_qkv_rope(
    const float* __restrict__ node,
    const float* __restrict__ Wq, const float* __restrict__ Wk, const float* __restrict__ Wv,
    bf16_t* __restrict__ qb, bf16_t* __restrict__ kb, bf16_t* __restrict__ vb) {
  __shared__ bf16_t sIn[16][272];
  __shared__ float  sOut[16][772];
  const int tid = threadIdx.x;
  const int lane = tid & 31, wv = tid >> 5;
  const int row0 = blockIdx.x * 16;

  for (int idx = tid; idx < 16 * 256; idx += 256) {
    int r = idx >> 8, c = idx & 255;
    sIn[r][c] = (bf16_t)node[(row0 + r) * 256 + c];
  }
  __syncthreads();

  for (int t = wv; t < 48; t += 8) {
    const float* W = (t < 16) ? Wq : (t < 32) ? Wk : Wv;
    const int ncol = (t & 15) * 16;
    v8f acc = {};
#pragma unroll
    for (int ks = 0; ks < 8; ++ks) {
      v16bf a = load_a_bf16(&sIn[0][0], 272, ks * 32, 16);
      v16bf b = load_b_f32(W + (ks * 32) * 256 + ncol, 256, 16);
      acc = wmma_bf16(a, b, acc);
    }
    const int nn = lane & 15, hi = lane >> 4;
#pragma unroll
    for (int r = 0; r < 8; ++r) sOut[r + 8 * hi][t * 16 + nn] = acc[r];
  }
  __syncthreads();

  for (int idx = tid; idx < 16 * 8 * 16; idx += 256) {
    int r = idx >> 7, rem = idx & 127;
    int hh = rem >> 4, t = rem & 15;
    int grow = row0 + r;
    int bb = grow >> 10, npos = grow & 1023;
    float ang = (float)npos * __powf(10000.0f, -(float)(2 * t) / 32.0f);
    float sn, cs;
    __sincosf(ang, &sn, &cs);
    long base = ((long)(bb * 8 + hh) * 1024 + npos) * 32;
    float q1 = sOut[r][hh * 32 + 2 * t], q2 = sOut[r][hh * 32 + 2 * t + 1];
    qb[base + 2 * t]     = (bf16_t)(q1 * cs - q2 * sn);
    qb[base + 2 * t + 1] = (bf16_t)(q1 * sn + q2 * cs);
    float k1 = sOut[r][256 + hh * 32 + 2 * t], k2 = sOut[r][256 + hh * 32 + 2 * t + 1];
    kb[base + 2 * t]     = (bf16_t)(k1 * cs - k2 * sn);
    kb[base + 2 * t + 1] = (bf16_t)(k1 * sn + k2 * cs);
  }
  for (int idx = tid; idx < 16 * 256; idx += 256) {
    int r = idx >> 8, c = idx & 255;
    int grow = row0 + r;
    int bb = grow >> 10, npos = grow & 1023;
    int hh = c >> 5, d = c & 31;
    vb[((long)(bb * 8 + hh) * 1024 + npos) * 32 + d] = (bf16_t)sOut[r][512 + c];
  }
}

// =============== Kernel 2: fused edge-bias + flash attention + P@V + P@E ===============
// TDM streams the 16 x 32 x 64 f32 edge block into LDS (row-padded to stride 65);
// wave w issues descriptors for rows {2w, 2w+1} and waits on its own TENSORcnt.
__global__ __launch_bounds__(256) void k_attn(
    const float* __restrict__ edge, const unsigned char* __restrict__ mask,
    const bf16_t* __restrict__ qb, const bf16_t* __restrict__ kb,
    const bf16_t* __restrict__ vb,
    const float* __restrict__ We, const float* __restrict__ be,
    float* __restrict__ X) {
  __shared__ float  sEf[16][32][65];    // TDM destination: 16 i x 32 j x 64 d (+1 pad)
  __shared__ bf16_t sBias[16][32][8];   // edge bias per (i,j,h)
  __shared__ bf16_t sP[8][16][34];      // softmaxed probs per (h,i,j)
  __shared__ float  sScale[8][16];
  __shared__ float  sL[8][16];
  __shared__ float  sBe[8];
  __shared__ unsigned char sMaskI[16];
  __shared__ unsigned char sMaskJ[32];

  const int tid = threadIdx.x;
  const int lane = tid & 31, wv = tid >> 5;
  const int nn = lane & 15, hi = lane >> 4;
  const int b = blockIdx.x >> 6;
  const int i0 = (blockIdx.x & 63) * 16;

  const float node_scale = 0.125f;          // (2*32)^-0.5
  const float edge_scale = 0.70710678118f;  // 2^-0.5

  if (tid < 8)  sBe[tid] = be[tid];
  if (tid < 16) sMaskI[tid] = mask[b * 1024 + i0 + tid];

  v16bf qfrag = load_a_bf16(qb + ((long)(b * 8 + wv) * 1024 + i0) * 32, 32, 0, 16);
  v16bf bWe0 = load_b_f32(We, 8, 8);
  v16bf bWe1 = load_b_f32(We + 32 * 8, 8, 8);

  float mrow[8], lrow[8];
#pragma unroll
  for (int r = 0; r < 8; ++r) { mrow[r] = NEG_MAX; lrow[r] = 0.0f; }
  v8f zero8 = {};
  v8f accN0 = zero8, accN1 = zero8;
  v8f accE[2][4];
#pragma unroll
  for (int s = 0; s < 2; ++s)
#pragma unroll
    for (int q = 0; q < 4; ++q) accE[s][q] = zero8;

  for (int j0 = 0; j0 < 1024; j0 += 32) {
    __syncthreads();  // previous iteration's reads of sEf/sP complete

    // ---- TDM: wave wv DMA-loads edge rows ii = {2wv, 2wv+1} into LDS ----
#pragma unroll
    for (int s = 0; s < 2; ++s) {
      int ii = wv * 2 + s;
      const float* gp = edge + ((long)(b * 1024 + i0 + ii) * 1024 + j0) * 64;
      tdm_load_rows_f32((unsigned)(uintptr_t)&sEf[ii][0][0], gp,
                        /*tile_rows=*/32, /*tensor_rows=*/1024 - (unsigned)j0,
                        /*row_stride=*/64);
    }
    if (tid < 32) sMaskJ[tid] = mask[b * 1024 + j0 + tid];
    __builtin_amdgcn_s_wait_tensorcnt(0);
    __syncthreads();  // all waves' tensor DMAs complete -> sEf valid block-wide

    // ---- edge bias GEMM: (i,j)-rows x d  @  We(d x h) ----
    for (int t = wv; t < 32; t += 8) {
      int ii = t >> 1, jh = t & 1;
      v8f d = {};
      d = wmma_bf16(load_a_f32(&sEf[ii][jh * 16][0], 65, 0, 16), bWe0, d);
      d = wmma_bf16(load_a_f32(&sEf[ii][jh * 16][0], 65, 32, 16), bWe1, d);
      if (nn < 8) {
#pragma unroll
        for (int r = 0; r < 8; ++r)
          sBias[ii][jh * 16 + r + 8 * hi][nn] = (bf16_t)(d[r] + sBe[nn]);
      }
    }
    __syncthreads();

    // ---- logits (Q @ K^T) + online softmax for head wv ----
    {
      const bf16_t* kbase = kb + ((long)(b * 8 + wv) * 1024 + j0) * 32;
      v8f s0 = wmma_bf16(qfrag, load_bT_bf16(kbase, 32), zero8);
      v8f s1 = wmma_bf16(qfrag, load_bT_bf16(kbase + 16 * 32, 32), zero8);
#pragma unroll
      for (int r = 0; r < 8; ++r) {
        int ii = r + 8 * hi;
        bool mi = sMaskI[ii] != 0;
        float lg0 = (mi && sMaskJ[nn])
                        ? s0[r] * node_scale + (float)sBias[ii][nn][wv] * edge_scale
                        : NEG_MAX;
        float lg1 = (mi && sMaskJ[16 + nn])
                        ? s1[r] * node_scale + (float)sBias[ii][16 + nn][wv] * edge_scale
                        : NEG_MAX;
        float mx = fmaxf(lg0, lg1);
#pragma unroll
        for (int off = 1; off < 16; off <<= 1) mx = fmaxf(mx, __shfl_xor(mx, off));
        float mnew = fmaxf(mrow[r], mx);
        float sc = __expf(mrow[r] - mnew);
        float p0 = __expf(lg0 - mnew);
        float p1 = __expf(lg1 - mnew);
        float rs = p0 + p1;
#pragma unroll
        for (int off = 1; off < 16; off <<= 1) rs += __shfl_xor(rs, off);
        lrow[r] = lrow[r] * sc + rs;
        mrow[r] = mnew;
        accN0[r] *= sc;
        accN1[r] *= sc;
        sP[wv][ii][nn] = (bf16_t)p0;
        sP[wv][ii][16 + nn] = (bf16_t)p1;
        if (nn == 0) sScale[wv][ii] = sc;
      }
    }
    __syncthreads();

    // ---- res_node: P(16x32) @ V(32x32) for head wv ----
    {
      v16bf pa = load_a_bf16(&sP[wv][0][0], 34, 0, 16);
      const bf16_t* vbase = vb + ((long)(b * 8 + wv) * 1024 + j0) * 32;
      accN0 = wmma_bf16(pa, load_b_bf16(vbase, 32), accN0);
      accN1 = wmma_bf16(pa, load_b_bf16(vbase + 16, 32), accN1);
    }
    // ---- res_edge: P(heads x 32j) @ E(32j x 64d) for owned rows ----
#pragma unroll
    for (int s = 0; s < 2; ++s) {
      int ii = wv * 2 + s;
      v16bf pe = load_a_bf16(&sP[0][ii][0], 16 * 34, 0, 8);  // rows = head, pad to 16
#pragma unroll
      for (int q = 0; q < 4; ++q) {
#pragma unroll
        for (int r = 0; r < 8; ++r) accE[s][q][r] *= sScale[(r + 8 * hi) & 7][ii];
        accE[s][q] = wmma_bf16(pe, load_b_f32(&sEf[ii][0][q * 16], 65, 16), accE[s][q]);
      }
    }
  }

  // ---- epilogue: normalize by l and write concat buffer X[b][i][0:768] ----
  if (nn == 0) {
#pragma unroll
    for (int r = 0; r < 8; ++r) sL[wv][r + 8 * hi] = lrow[r];
  }
#pragma unroll
  for (int r = 0; r < 8; ++r) {
    int ii = r + 8 * hi;
    float inv = 1.0f / lrow[r];
    long orow = (long)(b * 1024 + i0 + ii) * 768;
    X[orow + wv * 32 + nn] = accN0[r] * inv;
    X[orow + wv * 32 + 16 + nn] = accN1[r] * inv;
  }
  __syncthreads();
#pragma unroll
  for (int s = 0; s < 2; ++s) {
    int ii = wv * 2 + s;
    long orow = (long)(b * 1024 + i0 + ii) * 768;
#pragma unroll
    for (int q = 0; q < 4; ++q) {
#pragma unroll
      for (int r = 0; r < 8; ++r) {
        int m = r + 8 * hi;
        if (m < 8)
          X[orow + 256 + m * 64 + q * 16 + nn] = accE[s][q][r] / sL[m][ii];
      }
    }
  }
}

// =============== Kernel 3: output projection (2048x768)@(768x256)+bo ===============
__global__ __launch_bounds__(256) void k_outproj(
    const float* __restrict__ X, const float* __restrict__ Wo,
    const float* __restrict__ bo, float* __restrict__ out) {
  __shared__ bf16_t sX[16][784];
  const int tid = threadIdx.x;
  const int lane = tid & 31, wv = tid >> 5;
  const int nn = lane & 15, hi = lane >> 4;
  const int row0 = blockIdx.x * 16;

  for (int idx = tid; idx < 16 * 768; idx += 256) {
    int r = idx / 768, c = idx % 768;
    sX[r][c] = (bf16_t)X[(long)(row0 + r) * 768 + c];
  }
  __syncthreads();

  for (int t = wv; t < 16; t += 8) {
    v8f acc = {};
#pragma unroll
    for (int ks = 0; ks < 24; ++ks) {
      v16bf a = load_a_bf16(&sX[0][0], 784, ks * 32, 16);
      v16bf b = load_b_f32(Wo + (ks * 32) * 256 + t * 16, 256, 16);
      acc = wmma_bf16(a, b, acc);
    }
    float bias = bo[t * 16 + nn];
#pragma unroll
    for (int r = 0; r < 8; ++r)
      out[(long)(row0 + r + 8 * hi) * 256 + t * 16 + nn] = acc[r] + bias;
  }
}

// ==================================== launch ====================================
extern "C" void kernel_launch(void* const* d_in, const int* in_sizes, int n_in,
                              void* d_out, int out_size, void* d_ws, size_t ws_size,
                              hipStream_t stream) {
  const float* node = (const float*)d_in[0];
  const float* edge = (const float*)d_in[1];
  const unsigned char* mask = (const unsigned char*)d_in[2];
  const float* Wq = (const float*)d_in[3];
  const float* Wk = (const float*)d_in[4];
  const float* Wv = (const float*)d_in[5];
  const float* We = (const float*)d_in[6];
  const float* be = (const float*)d_in[7];
  const float* Wo = (const float*)d_in[8];
  const float* bo = (const float*)d_in[9];
  float* out = (float*)d_out;

  char* ws = (char*)d_ws;
  const size_t QKV = (size_t)2 * 8 * 1024 * 32;
  bf16_t* qb = (bf16_t*)ws;
  bf16_t* kb = qb + QKV;
  bf16_t* vb = kb + QKV;
  float* X = (float*)(ws + (size_t)3 * 1024 * 1024);  // 2*1024*768 f32 = 6MB

  k_qkv_rope<<<128, 256, 0, stream>>>(node, Wq, Wk, Wv, qb, kb, vb);
  k_attn<<<128, 256, 0, stream>>>(edge, mask, qb, kb, vb, We, be, X);
  k_outproj<<<128, 256, 0, stream>>>(X, Wo, bo, out);

  (void)in_sizes; (void)n_in; (void)out_size; (void)ws_size;
}
